// MOELayer_76828374991704
// MI455X (gfx1250) — matrix operand order repaired and checked by reading, not compiled, not run on hardware
//
#include <hip/hip_runtime.h>

// ---------------------------------------------------------------------------
// MoE Top-1 layer for MI455X (gfx1250, wave32, WMMA).
// T=2048 tokens, M=2048 model dim, E=8 experts, V=8192 hidden, CAP=256.
// Expert GEMMs run as bf16 WMMA (v_wmma_f32_16x16x32_bf16); f32 weights are
// converted to bf16 on the fly while staging through LDS so HBM traffic stays
// at the one mandatory pass over the 1.07 GB of weights (~46 us at 23.3 TB/s).
// Round 3: scalar-base + 32-bit-offset addressing for all staging/prefetch
// (saddr+voffset form; kills the per-iteration v_add_nc_u64 chains).
// ---------------------------------------------------------------------------

#define T_TOK 2048
#define M_DIM 2048
#define E_EXP 8
#define V_DIM 8192
#define CAPN  256

typedef __attribute__((ext_vector_type(16))) __bf16 v16bf;
typedef __attribute__((ext_vector_type(8)))  __bf16 v8bf;
typedef __attribute__((ext_vector_type(4)))  __bf16 v4bf;
typedef __attribute__((ext_vector_type(8)))  float  v8f;

// ------------------------- workspace layout (bytes) ------------------------
#define OFF_DISP   ((size_t)0)                      // bf16 [E*CAP, M]   8 MiB
#define OFF_H      ((size_t)8388608)                // bf16 [E*CAP, V]  32 MiB
#define OFF_EO     ((size_t)41943040)               // f32  [E*CAP, M]  16 MiB
#define OFF_EIDX   ((size_t)58720256)               // int  [T]
#define OFF_GATE   ((size_t)58728448)               // f32  [T]
#define OFF_SLOT   ((size_t)58736640)               // int  [T]
#define OFF_SCALE  ((size_t)58744832)               // f32  [T]
#define OFF_ME     ((size_t)58753024)               // f32  [E]

// ---------------------------------------------------------------------------
// 0) init: zero the dispatched buffer and the softmax-mass accumulators
// ---------------------------------------------------------------------------
__global__ void moe_init_kernel(__bf16* __restrict__ disp, float* __restrict__ me) {
    size_t i = (size_t)blockIdx.x * blockDim.x + threadIdx.x;   // 524288 v8bf chunks
    v8bf z = {};
    ((v8bf*)disp)[i] = z;
    if (blockIdx.x == 0 && threadIdx.x < E_EXP) me[threadIdx.x] = 0.0f;
}

// ---------------------------------------------------------------------------
// 1) gating: one wave per token. logits = x[t] . wg[e], softmax, argmax.
// ---------------------------------------------------------------------------
__global__ void moe_gate_kernel(const float* __restrict__ x,
                                const float* __restrict__ wg,
                                int* __restrict__ eidx,
                                float* __restrict__ gate,
                                float* __restrict__ me) {
    const int lane = threadIdx.x & 31;
    const int wave = threadIdx.x >> 5;
    const int t    = blockIdx.x * 8 + wave;

    const float* xr = x + (size_t)t * M_DIM;
    float acc[E_EXP];
#pragma unroll
    for (int e = 0; e < E_EXP; ++e) acc[e] = 0.0f;

    for (unsigned k = lane * 4; k < M_DIM; k += 32 * 4) {
        float4 xv = *(const float4*)(xr + k);
#pragma unroll
        for (int e = 0; e < E_EXP; ++e) {
            float4 wv = *(const float4*)(wg + e * M_DIM + k);
            acc[e] += xv.x * wv.x + xv.y * wv.y + xv.z * wv.z + xv.w * wv.w;
        }
    }
#pragma unroll
    for (int e = 0; e < E_EXP; ++e) {
#pragma unroll
        for (int off = 16; off > 0; off >>= 1)
            acc[e] += __shfl_xor(acc[e], off, 32);
    }
    if (lane == 0) {
        float mx = acc[0]; int am = 0;
#pragma unroll
        for (int e = 1; e < E_EXP; ++e)
            if (acc[e] > mx) { mx = acc[e]; am = e; }   // first-max tie-break
        float g[E_EXP]; float s = 0.0f;
#pragma unroll
        for (int e = 0; e < E_EXP; ++e) { g[e] = __expf(acc[e] - mx); s += g[e]; }
        float inv = 1.0f / s;
        eidx[t] = am;
        gate[t] = g[am] * inv;
#pragma unroll
        for (int e = 0; e < E_EXP; ++e) atomicAdd(&me[e], g[e] * inv);
    }
}

// ---------------------------------------------------------------------------
// 2) capacity scan: single wave. Per-expert ordered prefix count over tokens
//    via __shfl_up scan; emits slot/scale per token and l_aux.
// ---------------------------------------------------------------------------
__global__ void moe_scan_kernel(const int* __restrict__ eidx,
                                const float* __restrict__ gate,
                                const float* __restrict__ me,
                                int* __restrict__ slot,
                                float* __restrict__ scale,
                                float* __restrict__ laux_out) {
    const int lane = threadIdx.x;                   // 32 threads, 64 tokens each
    const int t0 = lane * (T_TOK / 32);

    int cnt[E_EXP];
#pragma unroll
    for (int e = 0; e < E_EXP; ++e) cnt[e] = 0;
    for (int i = 0; i < T_TOK / 32; ++i) cnt[eidx[t0 + i]]++;

    int run[E_EXP]; float tot[E_EXP];
#pragma unroll
    for (int e = 0; e < E_EXP; ++e) {
        int inc = cnt[e];
#pragma unroll
        for (int off = 1; off < 32; off <<= 1) {
            int n = __shfl_up(inc, off, 32);
            if (lane >= off) inc += n;
        }
        run[e] = inc - cnt[e];                      // exclusive prefix
        tot[e] = (float)__shfl(inc, 31, 32);        // ce[e]
    }
    for (int i = 0; i < T_TOK / 32; ++i) {
        int t = t0 + i;
        int e = eidx[t];
        int loc = run[e]++;
        bool valid = loc < CAPN;
        slot[t]  = valid ? (e * CAPN + loc) : -1;
        scale[t] = valid ? gate[t] : 0.0f;
    }
    if (lane == 0) {
        float l = 0.0f;
#pragma unroll
        for (int e = 0; e < E_EXP; ++e) l += me[e] * tot[e];
        *laux_out = l * ((float)E_EXP / ((float)T_TOK * (float)T_TOK));
    }
}

// ---------------------------------------------------------------------------
// 3) dispatch: gate-scaled scatter of x rows into bf16 [E*CAP, M]
// ---------------------------------------------------------------------------
__global__ void moe_dispatch_kernel(const float* __restrict__ x,
                                    const int* __restrict__ slot,
                                    const float* __restrict__ scale,
                                    __bf16* __restrict__ disp) {
    const int t = blockIdx.x;
    const int s = slot[t];
    if (s < 0) return;
    const float sc = scale[t];
    const int c0 = threadIdx.x * 8;
    const float4 a = *(const float4*)(x + (size_t)t * M_DIM + c0);
    const float4 b = *(const float4*)(x + (size_t)t * M_DIM + c0 + 4);
    v8bf o;
    o[0] = (__bf16)(sc * a.x); o[1] = (__bf16)(sc * a.y);
    o[2] = (__bf16)(sc * a.z); o[3] = (__bf16)(sc * a.w);
    o[4] = (__bf16)(sc * b.x); o[5] = (__bf16)(sc * b.y);
    o[6] = (__bf16)(sc * b.z); o[7] = (__bf16)(sc * b.w);
    *(v8bf*)(disp + (size_t)s * M_DIM + c0) = o;
}

// ---------------------------------------------------------------------------
// 4/5) expert GEMM: Out[e] = act(A[e] (bf16, CxK) * W[e] (f32->bf16, KxN) + b)
//      128x128 block tile, BLK_K=64, 8 waves of 64x32, WMMA f32_16x16x32_bf16.
//      All staging addresses are uniform-base + 32-bit offsets (saddr form).
// ---------------------------------------------------------------------------
#define BLK_M 128
#define BLK_N 128
#define BLK_K 64
#define LDS_STRIDE 72   // padded (bank stride 36 -> conflict-free half-wave reads)

__device__ __forceinline__ float f4_comp(const float4& v, int j) {
    switch (j) { case 0: return v.x; case 1: return v.y; case 2: return v.z;
                 default: return v.w; }
}

template <bool RELU_BF16_OUT>
__global__ __launch_bounds__(256, 1)
void expert_gemm_kernel(const __bf16* __restrict__ A,   // [E, C, K]
                        const float*  __restrict__ W,   // [E, K, N]
                        const float*  __restrict__ bias,// [E, N]
                        void*         __restrict__ Out, // [E, C, N]
                        int K, int N) {
    __shared__ __attribute__((aligned(32))) __bf16 As[BLK_M * LDS_STRIDE];
    __shared__ __attribute__((aligned(32))) __bf16 Bs[BLK_N * LDS_STRIDE]; // [n][k]

    const int tid   = threadIdx.x;
    const int lane  = tid & 31;
    const int wave  = tid >> 5;
    const int waveM = wave >> 2;          // 0..1  (64-row strip)
    const int waveN = wave & 3;           // 0..3  (32-col strip)
    const int e     = blockIdx.z;
    const int mbase = blockIdx.y * BLK_M;
    const int nbase = blockIdx.x * BLK_N;

    // uniform scalar bases; per-expert arrays are < 2 GiB so 32-bit offsets ok
    const __bf16* Ae = A + (size_t)e * CAPN * K + (size_t)mbase * K;  // tile row 0
    const float*  We = W + (size_t)e * K * N + nbase;                 // tile col 0

    v8f acc[4][2];
#pragma unroll
    for (int mi = 0; mi < 4; ++mi)
#pragma unroll
        for (int ni = 0; ni < 2; ++ni) acc[mi][ni] = (v8f)0.0f;

    const int lhalf = lane & 15;
    const int aChunkBase = (lane < 16) ? 0 : 8;    // halves
    const int bKoff      = (lane < 16) ? 0 : 8;    // halves

    // per-thread staging coordinates (loop-invariant 32-bit element offsets)
    const unsigned aRow = tid >> 3;                  // A: row, k-chunk
    const unsigned aKc  = tid & 7;
    const unsigned aOff = aRow * (unsigned)K + aKc * 8;
    const unsigned bK4  = (tid >> 5) * 4;            // B: 4x4 block
    const unsigned bN4  = (tid & 31) * 4;
    const unsigned bOff = bK4 * (unsigned)N + bN4;

    for (int kk = 0; kk < K; kk += BLK_K) {
        const unsigned kku  = (unsigned)kk;
        const unsigned kkNu = kku * (unsigned)N;
        // ---- stage A tile: 128 x 64 bf16, 4 x 16B chunks per thread -------
#pragma unroll
        for (int i = 0; i < 4; ++i) {
            v8bf v = *(const v8bf*)(Ae + (aOff + (unsigned)(i * 32) * (unsigned)K + kku));
            *(v8bf*)(&As[(aRow + i * 32) * LDS_STRIDE + aKc * 8]) = v;
        }
        // ---- stage W tile transposed [n][k]: 4x4 blocks, b64 LDS stores ---
#pragma unroll
        for (int i = 0; i < 2; ++i) {
            const unsigned blkOff = bOff + (unsigned)(i * 32) * (unsigned)N + kkNu;
            float4 r[4];
#pragma unroll
            for (int rr = 0; rr < 4; ++rr)
                r[rr] = *(const float4*)(We + (blkOff + (unsigned)rr * (unsigned)N));
            const int k4 = (int)bK4 + i * 32;
#pragma unroll
            for (int j = 0; j < 4; ++j) {
                v4bf p;
                p[0] = (__bf16)f4_comp(r[0], j);
                p[1] = (__bf16)f4_comp(r[1], j);
                p[2] = (__bf16)f4_comp(r[2], j);
                p[3] = (__bf16)f4_comp(r[3], j);
                *(v4bf*)(&Bs[(bN4 + j) * LDS_STRIDE + k4]) = p;
            }
        }
        __syncthreads();

        // ---- prefetch next K-tile into cache while WMMAs run --------------
        if (kk + BLK_K < K) {
            const unsigned pk  = (unsigned)(tid >> 2);   // 64 rows of W
            const unsigned seg = (unsigned)(tid & 3);    // 4 x 128B lines per row
            __builtin_prefetch(We + (kkNu + (unsigned)(BLK_K * N) + pk * (unsigned)N + seg * 32), 0, 3);
            if (tid < BLK_M)                             // 128 A rows, 1 line each
                __builtin_prefetch(Ae + ((unsigned)tid * (unsigned)K + kku + BLK_K), 0, 3);
        }

        // ---- compute: 2 k-slices of 32, 4x2 tiles per wave ---------------
#pragma unroll
        for (int s = 0; s < 2; ++s) {
            const int kb = s * 32;                 // halves
            v16bf af[4], bfr[2];
#pragma unroll
            for (int mi = 0; mi < 4; ++mi) {
                int r = waveM * 64 + mi * 16 + lhalf;
                v8bf lo = *(const v8bf*)(&As[r * LDS_STRIDE + kb + aChunkBase]);
                v8bf hi = *(const v8bf*)(&As[r * LDS_STRIDE + kb + aChunkBase + 16]);
                af[mi] = __builtin_shufflevector(lo, hi, 0, 1, 2, 3, 4, 5, 6, 7,
                                                 8, 9, 10, 11, 12, 13, 14, 15);
            }
#pragma unroll
            for (int ni = 0; ni < 2; ++ni) {
                int cn = waveN * 32 + ni * 16 + lhalf;
                v8bf lo = *(const v8bf*)(&Bs[cn * LDS_STRIDE + kb + bKoff]);
                v8bf hi = *(const v8bf*)(&Bs[cn * LDS_STRIDE + kb + bKoff + 16]);
                bfr[ni] = __builtin_shufflevector(lo, hi, 0, 1, 2, 3, 4, 5, 6, 7,
                                                  8, 9, 10, 11, 12, 13, 14, 15);
            }
#pragma unroll
            for (int mi = 0; mi < 4; ++mi)
#pragma unroll
                for (int ni = 0; ni < 2; ++ni)
                    acc[mi][ni] = __builtin_amdgcn_wmma_f32_16x16x32_bf16(
                        false, af[mi], false, bfr[ni], (short)0, acc[mi][ni],
                        false, false);
        }
        __syncthreads();
    }

    // ---- epilogue: bias (+ReLU), scatter per D-matrix layout --------------
    float bv[2];
#pragma unroll
    for (int ni = 0; ni < 2; ++ni)
        bv[ni] = bias[(size_t)e * N + nbase + waveN * 32 + ni * 16 + lhalf];

    const int mrow0 = mbase + waveM * 64 + ((lane < 16) ? 0 : 8);
#pragma unroll
    for (int mi = 0; mi < 4; ++mi) {
#pragma unroll
        for (int ni = 0; ni < 2; ++ni) {
            int n = nbase + waveN * 32 + ni * 16 + lhalf;
#pragma unroll
            for (int i = 0; i < 8; ++i) {
                float v = acc[mi][ni][i] + bv[ni];
                if (RELU_BF16_OUT) v = fmaxf(v, 0.0f);
                int m = mrow0 + mi * 16 + i;
                size_t off = ((size_t)e * CAPN + m) * (size_t)N + n;
                if (RELU_BF16_OUT) ((__bf16*)Out)[off] = (__bf16)v;
                else               ((float*)Out)[off] = v;
            }
        }
    }
}

// ---------------------------------------------------------------------------
// 6) combine: out[t] = scale[t] * eo[slot[t]]  (zeros for dropped tokens)
// ---------------------------------------------------------------------------
__global__ void moe_combine_kernel(const float* __restrict__ eo,
                                   const int* __restrict__ slot,
                                   const float* __restrict__ scale,
                                   float* __restrict__ out) {
    const int t = blockIdx.x;
    const int s = slot[t];
    const int c0 = threadIdx.x * 8;
    float4 r0 = make_float4(0.f, 0.f, 0.f, 0.f);
    float4 r1 = r0;
    if (s >= 0) {
        const float sc = scale[t];
        float4 a = *(const float4*)(eo + (size_t)s * M_DIM + c0);
        float4 b = *(const float4*)(eo + (size_t)s * M_DIM + c0 + 4);
        r0 = make_float4(sc * a.x, sc * a.y, sc * a.z, sc * a.w);
        r1 = make_float4(sc * b.x, sc * b.y, sc * b.z, sc * b.w);
    }
    *(float4*)(out + (size_t)t * M_DIM + c0)     = r0;
    *(float4*)(out + (size_t)t * M_DIM + c0 + 4) = r1;
}

// ---------------------------------------------------------------------------
extern "C" void kernel_launch(void* const* d_in, const int* in_sizes, int n_in,
                              void* d_out, int out_size, void* d_ws, size_t ws_size,
                              hipStream_t stream) {
    (void)in_sizes; (void)n_in; (void)out_size; (void)ws_size;
    const float* x     = (const float*)d_in[0];
    const float* wg    = (const float*)d_in[1];
    const float* fc1_w = (const float*)d_in[2];
    const float* fc1_b = (const float*)d_in[3];
    const float* fc2_w = (const float*)d_in[4];
    const float* fc2_b = (const float*)d_in[5];
    float* out = (float*)d_out;

    char* ws = (char*)d_ws;
    __bf16* disp  = (__bf16*)(ws + OFF_DISP);
    __bf16* h     = (__bf16*)(ws + OFF_H);
    float*  eo    = (float*)(ws + OFF_EO);
    int*    eidx  = (int*)(ws + OFF_EIDX);
    float*  gate  = (float*)(ws + OFF_GATE);
    int*    slot  = (int*)(ws + OFF_SLOT);
    float*  scale = (float*)(ws + OFF_SCALE);
    float*  me    = (float*)(ws + OFF_ME);

    // 0) zero dispatched buffer (524288 x 16B) + me
    moe_init_kernel<<<2048, 256, 0, stream>>>(disp, me);
    // 1) gating: 8 tokens per block (one wave each)
    moe_gate_kernel<<<T_TOK / 8, 256, 0, stream>>>(x, wg, eidx, gate, me);
    // 2) capacity scan + l_aux (written at out[T*M])
    moe_scan_kernel<<<1, 32, 0, stream>>>(eidx, gate, me, slot, scale,
                                          out + (size_t)T_TOK * M_DIM);
    // 3) dispatch to bf16
    moe_dispatch_kernel<<<T_TOK, 256, 0, stream>>>(x, slot, scale, disp);
    // 4) fc1: [E,256,2048] x [E,2048,8192] -> relu -> bf16 h
    expert_gemm_kernel<true><<<dim3(V_DIM / BLK_N, CAPN / BLK_M, E_EXP), 256, 0, stream>>>(
        disp, fc1_w, fc1_b, (void*)h, M_DIM, V_DIM);
    // 5) fc2: [E,256,8192] x [E,8192,2048] -> f32 eo
    expert_gemm_kernel<false><<<dim3(M_DIM / BLK_N, CAPN / BLK_M, E_EXP), 256, 0, stream>>>(
        h, fc2_w, fc2_b, (void*)eo, V_DIM, M_DIM);
    // 6) gather/combine
    moe_combine_kernel<<<T_TOK, 256, 0, stream>>>(eo, slot, scale, out);
}
